// LineFinder_11201274708550
// MI455X (gfx1250) — matrix operand order repaired and checked by reading, not compiled, not run on hardware
//
#include <hip/hip_runtime.h>

// ---------------------------------------------------------------------------
// ResNet50 front-end (conv1..layer2 + head) for MI455X / gfx1250.
// All 1x1/3x3 convs -> one WMMA implicit-GEMM kernel (V_WMMA_F32_16X16X32_BF16,
// bf16 in / fp32 accumulate). Software-pipelined: global->regs issued one
// K-step ahead, regs->LDS behind a single barrier per step, double-buffered
// LDS, pure ds_load_b128 fragments. Wave tile 32x64 (2x4 WMMA) for higher
// WMMA : LDS-read ratio.
// ---------------------------------------------------------------------------

typedef unsigned short u16;
typedef __attribute__((ext_vector_type(16))) __bf16 v16bf;
typedef __attribute__((ext_vector_type(8)))  float  v8f;

union ABu { unsigned u[8]; v16bf v; };   // 32B-aligned via v16bf member

__device__ __forceinline__ float bf2f(u16 h) {
  unsigned u = ((unsigned)h) << 16;
  float f;
  __builtin_memcpy(&f, &u, 4);
  return f;
}
__device__ __forceinline__ u16 f2bf(float f) {
  unsigned u;
  __builtin_memcpy(&u, &f, 4);
  u += 0x7FFFu + ((u >> 16) & 1u);   // round-to-nearest-even
  return (u16)(u >> 16);
}

// ---------------------------------------------------------------------------
__global__ void cvt_f32_bf16_kernel(const float* __restrict__ in,
                                    u16* __restrict__ out, int n) {
  int t = blockIdx.x * 256 + threadIdx.x;
  if (t < n) out[t] = f2bf(in[t]);
}

// ---------------------------------------------------------------------------
// conv1: 7x7 s2 p3, 3->64ch, BN+ReLU. fp32 in, bf16 out [8,64,256,256].
// ---------------------------------------------------------------------------
__global__ __launch_bounds__(256)
void conv1_bn_relu_kernel(const float* __restrict__ x, const float* __restrict__ w,
                          const float* __restrict__ s, const float* __restrict__ bi,
                          u16* __restrict__ out) {
  int t = blockIdx.x * 256 + threadIdx.x;
  const int ow = t & 255;
  const int oh = (t >> 8) & 255;
  const int oc = (t >> 16) & 63;
  const int b  = t >> 22;
  float acc = 0.0f;
  const float* wp = w + oc * 3 * 49;
  for (int c = 0; c < 3; ++c) {
    const float* xp = x + ((size_t)b * 3 + c) * 512 * 512;
    for (int kh = 0; kh < 7; ++kh) {
      int ih = oh * 2 - 3 + kh;
      if ((unsigned)ih >= 512u) continue;
      for (int kw = 0; kw < 7; ++kw) {
        int iw = ow * 2 - 3 + kw;
        if ((unsigned)iw >= 512u) continue;
        acc = __builtin_fmaf(xp[ih * 512 + iw], wp[c * 49 + kh * 7 + kw], acc);
      }
    }
  }
  out[t] = f2bf(fmaxf(acc * s[oc] + bi[oc], 0.0f));
}

// ---------------------------------------------------------------------------
// maxpool 3x3 s2 p1: [8,64,256,256] -> [8,64,128,128], bf16
// ---------------------------------------------------------------------------
__global__ __launch_bounds__(256)
void maxpool_kernel(const u16* __restrict__ in, u16* __restrict__ out) {
  int t = blockIdx.x * 256 + threadIdx.x;
  const int ow = t & 127;
  const int oh = (t >> 7) & 127;
  const int c  = (t >> 14) & 63;
  const int b  = t >> 20;
  const u16* p = in + ((size_t)(b * 64 + c)) * 65536;
  float m = -3.4e38f;
  for (int i = 0; i < 3; ++i) {
    int ih = oh * 2 - 1 + i;
    if ((unsigned)ih >= 256u) continue;
    for (int j = 0; j < 3; ++j) {
      int iw = ow * 2 - 1 + j;
      if ((unsigned)iw >= 256u) continue;
      m = fmaxf(m, bf2f(p[ih * 256 + iw]));
    }
  }
  out[t] = f2bf(m);
}

// ---------------------------------------------------------------------------
// WMMA implicit-GEMM conv, fused BN (+residual) (+ReLU).
//   M=C_out, N=8*Hout*Wout, K=C_in*ks*ks. 256 thr (8 wave32), 64x256 tile,
//   K step 32, double-buffered LDS, global loads pipelined one step ahead.
//   Wave grid 2(M)x4(N); each wave 32x64 => 2x4 WMMA per K-step.
//   As[buf]: [m][k] stride 32;  Bs[buf]: [n][k] stride 40 (16B-friendly pad).
// ---------------------------------------------------------------------------
#define TM 64
#define TN 256
#define BSTR 40

__global__ __launch_bounds__(256)
void wmma_conv_kernel(const u16* __restrict__ X, const u16* __restrict__ Wt,
                      const float* __restrict__ sc, const float* __restrict__ bi,
                      const u16* __restrict__ res, u16* __restrict__ out,
                      int Cin, int Hin, int Win, int Cout,
                      int ks, int stride, int pad, int do_relu, int lgW) {
  __shared__ __align__(16) u16 As[2][TM * 32];    //  8 KB
  __shared__ __align__(16) u16 Bs[2][TN * BSTR];  // 40 KB

  const int tid  = threadIdx.x;
  const int lane = tid & 31;
  const int wave = tid >> 5;
  const int l16  = lane & 15;
  const int half = lane >> 4;
  const int wm   = wave >> 2;         // 0..1 (M, 32 rows each)
  const int wn   = wave & 3;          // 0..3 (N, 64 cols each)
  const int m0   = blockIdx.y * TM;
  const int n0   = blockIdx.x * TN;
  const int K    = Cin * ks * ks;
  const int lgHW = lgW << 1;          // Hout==Wout, powers of two
  const int HWo  = 1 << lgHW;
  const int Wout = 1 << lgW;
  const int HWin = Hin * Win;

  // loader coordinates
  const int am  = tid >> 2;           // A row 0..63
  const int akk = (tid & 3) * 8;      // A k-offset
  const int kp  = tid >> 4;           // B k-pair 0..15
  const int k0  = kp * 2;
  const int nb  = (tid & 15) * 16;    // B n-offset 0..240
  const bool fast = (ks == 1) && (stride == 1);   // pure GEMM path

  // pipeline registers (filled one K-step ahead of LDS store)
  uint4 aReg;
  union { uint4 q[4]; u16 h[32]; } bReg;  // h[0..15]=k0 plane, h[16..31]=k0+1

  // ---- phase 1: global -> registers for K-step `kt`
  auto gload = [&](int kt) {
    {
      const u16* src = Wt + (size_t)(m0 + am) * K + (kt + akk);
      aReg = *(const uint4*)src;
      if (kt + 32 < K) __builtin_prefetch(src + 32, 0, 1);  // global_prefetch_b8
    }
    const int kg0 = kt + k0;
    if (fast) {
      const int ng0 = n0 + nb;            // multiple of 16, stays in one batch
      const int b   = ng0 >> lgHW;
      const int sp  = ng0 & (HWo - 1);
      const u16* p0 = X + ((size_t)b * Cin + kg0) * HWo + sp;
      bReg.q[0] = *(const uint4*)p0;              // cin = kg0,   n 0..7
      bReg.q[1] = *(const uint4*)(p0 + 8);        // cin = kg0,   n 8..15
      bReg.q[2] = *(const uint4*)(p0 + HWo);      // cin = kg0+1, n 0..7
      bReg.q[3] = *(const uint4*)(p0 + HWo + 8);  // cin = kg0+1, n 8..15
    } else {
      int cin0, kh0, kw0, cin1, kh1, kw1;
      if (ks == 3) {                      // constant-divisor decompose
        cin0 = kg0 / 9; int r0 = kg0 - cin0 * 9; kh0 = r0 / 3; kw0 = r0 - kh0 * 3;
        const int kg1 = kg0 + 1;
        cin1 = kg1 / 9; int r1 = kg1 - cin1 * 9; kh1 = r1 / 3; kw1 = r1 - kh1 * 3;
      } else {                            // 1x1 stride2 (downsample)
        cin0 = kg0; kh0 = 0; kw0 = 0; cin1 = kg0 + 1; kh1 = 0; kw1 = 0;
      }
      const size_t cb0 = (size_t)cin0 * HWin;
      const size_t cb1 = (size_t)cin1 * HWin;
#pragma unroll
      for (int e = 0; e < 16; ++e) {
        const int ng = n0 + nb + e;
        const int b  = ng >> lgHW;
        const int sp = ng & (HWo - 1);
        const int oh = sp >> lgW;
        const int ow = sp & (Wout - 1);
        const size_t bbase = (size_t)b * Cin * HWin;
        const int ih0 = oh * stride - pad + kh0;
        const int iw0 = ow * stride - pad + kw0;
        const int ih1 = oh * stride - pad + kh1;
        const int iw1 = ow * stride - pad + kw1;
        // branchless: clamp address (always-valid load) + zero-select
        const int ih0c = min(max(ih0, 0), Hin - 1);
        const int iw0c = min(max(iw0, 0), Win - 1);
        const int ih1c = min(max(ih1, 0), Hin - 1);
        const int iw1c = min(max(iw1, 0), Win - 1);
        const u16 v0 = X[bbase + cb0 + (size_t)ih0c * Win + iw0c];
        const u16 v1 = X[bbase + cb1 + (size_t)ih1c * Win + iw1c];
        const bool ok0 = ((unsigned)ih0 < (unsigned)Hin) & ((unsigned)iw0 < (unsigned)Win);
        const bool ok1 = ((unsigned)ih1 < (unsigned)Hin) & ((unsigned)iw1 < (unsigned)Win);
        bReg.h[e]      = ok0 ? v0 : (u16)0;
        bReg.h[16 + e] = ok1 ? v1 : (u16)0;
      }
    }
  };

  // ---- phase 2: registers -> LDS (waits on loads issued a full step earlier)
  auto storeTile = [&](int buf) {
    *(uint4*)&As[buf][am * 32 + akk] = aReg;
#pragma unroll
    for (int e = 0; e < 16; ++e) {
      unsigned pk = (unsigned)bReg.h[e] | ((unsigned)bReg.h[16 + e] << 16);
      *(unsigned*)&Bs[buf][(nb + e) * BSTR + k0] = pk;
    }
  };

  v8f acc[2][4];
  {
    v8f z = {0.f, 0.f, 0.f, 0.f, 0.f, 0.f, 0.f, 0.f};
#pragma unroll
    for (int i = 0; i < 2; ++i)
#pragma unroll
      for (int j = 0; j < 4; ++j) acc[i][j] = z;
  }

  gload(0);
  for (int kt = 0;; kt += 32) {
    const int  cb   = (kt >> 5) & 1;
    const bool more = (kt + 32) < K;
    storeTile(cb);                 // consumes regs loaded last iteration
    if (more) gload(kt + 32);      // issue next global loads before barrier
    __syncthreads();               // tile[cb] visible to all waves

    ABu a[2], bfr[4];
#pragma unroll
    for (int i = 0; i < 2; ++i) {
      // A 16x32 bf16 frag: K groups {half*8..+7, 16+half*8..+7} => 2x b128
      const u16* ap = &As[cb][(wm * 32 + i * 16 + l16) * 32 + half * 8];
      *(uint4*)&a[i].u[0] = *(const uint4*)ap;
      *(uint4*)&a[i].u[4] = *(const uint4*)(ap + 16);
    }
#pragma unroll
    for (int j = 0; j < 4; ++j) {
      // B 32x16 bf16 frag: K groups {half*8..+7, 16+half*8..+7} => 2x b128
      const u16* bp = &Bs[cb][(wn * 64 + j * 16 + l16) * BSTR + half * 8];
      *(uint4*)&bfr[j].u[0] = *(const uint4*)bp;
      *(uint4*)&bfr[j].u[4] = *(const uint4*)(bp + 16);
    }
#pragma unroll
    for (int i = 0; i < 2; ++i)
#pragma unroll
      for (int j = 0; j < 4; ++j)
        acc[i][j] = __builtin_amdgcn_wmma_f32_16x16x32_bf16(
            false, a[i].v, false, bfr[j].v, (short)0, acc[i][j], false, false);

    if (!more) break;
  }

  // ---- epilogue: BN scale/bias, optional residual, optional ReLU
#pragma unroll
  for (int i = 0; i < 2; ++i) {
    const int mb = m0 + wm * 32 + i * 16 + half * 8;   // C/D: M = r + 8*half
#pragma unroll
    for (int j = 0; j < 4; ++j) {
      const int n  = n0 + wn * 64 + j * 16 + l16;
      const int b  = n >> lgHW;
      const int sp = n & (HWo - 1);
#pragma unroll
      for (int r = 0; r < 8; ++r) {
        const int m = mb + r;
        float v = acc[i][j][r] * sc[m] + bi[m];
        const size_t idx = ((size_t)b * Cout + m) * (size_t)HWo + sp;
        if (res) v += bf2f(res[idx]);
        if (do_relu) v = fmaxf(v, 0.0f);
        out[idx] = f2bf(v);
      }
    }
  }
}

// ---------------------------------------------------------------------------
// head: 1x1 conv 512->4 + bias, + row/col offsets, sigmoid ch3,
// transpose to [B, h*w, 4] fp32. act: bf16 [8,512,64,64]
// ---------------------------------------------------------------------------
__global__ __launch_bounds__(128)
void final_head_kernel(const u16* __restrict__ act, const float* __restrict__ w,
                       const float* __restrict__ bias, float* __restrict__ out) {
  int t = blockIdx.x * 128 + threadIdx.x;   // 0..32767
  const int b  = t >> 12;
  const int sp = t & 4095;
  float a0 = bias[0], a1 = bias[1], a2 = bias[2], a3 = bias[3];
  const u16* ap = act + ((size_t)b * 512) * 4096 + sp;
  for (int k = 0; k < 512; ++k) {
    float a = bf2f(ap[(size_t)k * 4096]);
    a0 = __builtin_fmaf(a, w[k],        a0);
    a1 = __builtin_fmaf(a, w[512 + k],  a1);
    a2 = __builtin_fmaf(a, w[1024 + k], a2);
    a3 = __builtin_fmaf(a, w[1536 + k], a3);
  }
  const int row = sp >> 6, col = sp & 63;
  a0 += 4.0f + 8.0f * (float)row;
  a1 += 4.0f + 8.0f * (float)col;
  a3 = 1.0f / (1.0f + expf(-a3));
  float* o = out + (size_t)t * 4;
  o[0] = a0; o[1] = a1; o[2] = a2; o[3] = a3;
}

// ---------------------------------------------------------------------------
// host orchestration
// ---------------------------------------------------------------------------
extern "C" void kernel_launch(void* const* d_in, const int* in_sizes, int n_in,
                              void* d_out, int out_size, void* d_ws, size_t ws_size,
                              hipStream_t stream) {
  (void)in_sizes; (void)n_in; (void)out_size; (void)ws_size;

  const float* x       = (const float*)d_in[0];
  const float* conv1_w = (const float*)d_in[1];
  const float* bn1_s   = (const float*)d_in[2];
  const float* bn1_b   = (const float*)d_in[3];

  // ---- workspace carve (256B aligned)
  char*  ws  = (char*)d_ws;
  size_t off = 0;
  auto carve = [&](size_t bytes) -> void* {
    void* p = ws + off;
    off += (bytes + 255) & ~(size_t)255;
    return p;
  };
  u16* wbf  = (u16*)carve(2 * 1441792ull);    // bf16 weights
  u16* bufA = (u16*)carve(2 * 33554432ull);
  u16* bufB = (u16*)carve(2 * 33554432ull);
  u16* bufC = (u16*)carve(2 * 16777216ull);
  u16* bufD = (u16*)carve(2 * 8388608ull);

  size_t woff = 0;
  auto cvtW = [&](int inIdx, size_t n) -> const u16* {
    u16* dst = wbf + woff;
    woff += n;                                // all sizes multiples of 8
    cvt_f32_bf16_kernel<<<(unsigned)((n + 255) / 256), 256, 0, stream>>>(
        (const float*)d_in[inIdx], dst, (int)n);
    return dst;
  };
  auto S = [&](int i) -> const float* { return (const float*)d_in[i]; };

  auto conv = [&](const u16* Xp, const u16* Wp, const float* sc, const float* bi,
                  const u16* resp, u16* outp, int Cin, int Hin, int Cout,
                  int Hout, int ks, int stride, int pad, bool relu) {
    const int lgW = 31 - __builtin_clz((unsigned)Hout);
    dim3 grid((unsigned)((8 * Hout * Hout) / TN), (unsigned)(Cout / TM));
    wmma_conv_kernel<<<grid, 256, 0, stream>>>(
        Xp, Wp, sc, bi, resp, outp, Cin, Hin, Hin, Cout,
        ks, stride, pad, relu ? 1 : 0, lgW);
  };

  // ---- stem
  conv1_bn_relu_kernel<<<33554432u / 256u, 256, 0, stream>>>(
      x, conv1_w, bn1_s, bn1_b, bufA);
  maxpool_kernel<<<8388608u / 256u, 256, 0, stream>>>(bufA, bufB);

  u16* cur   = bufB;
  u16* other = bufA;
  int  ii    = 4;

  auto runBlock = [&](int cin, int mid, int cout, int Hin, int stride, bool down) {
    int w1 = ii++, s1 = ii++, b1 = ii++;
    int w2 = ii++, s2 = ii++, b2 = ii++;
    int w3 = ii++, s3 = ii++, b3 = ii++;
    int wd = -1, sd = -1, bd = -1;
    if (down) { wd = ii++; sd = ii++; bd = ii++; }
    const int Hout = Hin / stride;

    const u16* w1p = cvtW(w1, (size_t)mid * cin);
    conv(cur, w1p, S(s1), S(b1), nullptr, bufC, cin, Hin, mid, Hin, 1, 1, 0, true);

    const u16* w2p = cvtW(w2, (size_t)mid * mid * 9);
    conv(bufC, w2p, S(s2), S(b2), nullptr, bufD, mid, Hin, mid, Hout, 3, stride, 1, true);

    const u16* w3p = cvtW(w3, (size_t)cout * mid);
    if (down) {
      const u16* wdp = cvtW(wd, (size_t)cout * cin);
      conv(cur, wdp, S(sd), S(bd), nullptr, other, cin, Hin, cout, Hout, 1, stride, 0, false);
      conv(bufD, w3p, S(s3), S(b3), other, cur, mid, Hout, cout, Hout, 1, 1, 0, true);
    } else {
      conv(bufD, w3p, S(s3), S(b3), cur, other, mid, Hout, cout, Hout, 1, 1, 0, true);
      u16* t = cur; cur = other; other = t;
    }
  };

  // layer1: 64 -> 256 @128
  runBlock(64, 64, 256, 128, 1, true);
  runBlock(256, 64, 256, 128, 1, false);
  runBlock(256, 64, 256, 128, 1, false);
  // layer2: 256 -> 512, 128 -> 64
  runBlock(256, 128, 512, 128, 2, true);
  runBlock(512, 128, 512, 64, 1, false);
  runBlock(512, 128, 512, 64, 1, false);
  runBlock(512, 128, 512, 64, 1, false);

  final_head_kernel<<<32768u / 128u, 128, 0, stream>>>(
      cur, (const float*)d_in[73], (const float*)d_in[74], (float*)d_out);
}
